// Head_16630113370574
// MI455X (gfx1250) — matrix operand order repaired and checked by reading, not compiled, not run on hardware
//
#include <hip/hip_runtime.h>
#include <hip/hip_bf16.h>
#include <math.h>

typedef __attribute__((ext_vector_type(16))) _Float16 v16h;
typedef __attribute__((ext_vector_type(8)))  _Float16 v8h;
typedef __attribute__((ext_vector_type(8)))  float    v8f;
typedef __attribute__((ext_vector_type(4)))  unsigned int u32x4;
typedef __attribute__((ext_vector_type(8)))  int      i32x8;
typedef __attribute__((ext_vector_type(4)))  int      i32x4;

#define BATCH 8
#define SEQ   4096
#define EMB   1024
#define HEAD  128

// ---- CDNA5 async-copy helpers (ASYNCcnt-tracked, CDNA5 ISA) ----
__device__ __forceinline__ unsigned int lds_addr32(const void* p) {
    // flat LDS aperture: addr[31:0] is the LDS byte address (ISA 10.2)
    return (unsigned int)(unsigned long long)p;
}
__device__ __forceinline__ void async_copy_b128(unsigned int lds, const void* g) {
    asm volatile("global_load_async_to_lds_b128 %0, %1, off"
                 :: "v"(lds), "v"((unsigned long long)g) : "memory");
}
__device__ __forceinline__ void wait_async0() {
    asm volatile("s_wait_asynccnt 0x0" ::: "memory");
}

// ---- Tensor Data Mover: 2-D f16 tile load, D# per ISA 08 §8.3-8.4 ----------
// tile = tile_rows x HEAD f16, row-major, tensor_dim0_stride = HEAD.
__device__ __forceinline__ void tdm_load_tile_f16(unsigned int lds,
                                                  const void* gaddr,
                                                  unsigned int tile_rows) {
    unsigned long long ga = (unsigned long long)gaddr;
    u32x4 g0;
    g0[0] = 1u;                                      // count=1, no gather
    g0[1] = lds;                                     // lds_addr [63:32]
    g0[2] = (unsigned int)ga;                        // global_addr low
    g0[3] = (unsigned int)(ga >> 32) | (2u << 30);   // addr[56:32] | type=2
    i32x8 g1;
    g1[0] = (int)(1u << 16);          // data_size=1 (2 bytes), mask=0
    g1[1] = (int)(HEAD << 16);        // tensor_dim0=128 (low16 @ bits63:48)
    g1[2] = (int)(SEQ  << 16);        // tensor_dim1 low16 @ bits95:80
    g1[3] = (int)(HEAD << 16);        // tile_dim0=128 @ bits127:112
    g1[4] = (int)tile_rows;           // tile_dim1 @ bits143:128
    g1[5] = (int)HEAD;                // tensor_dim0_stride=128 (low32)
    g1[6] = 0;
    g1[7] = 0;
    i32x4 z4 = {0, 0, 0, 0};
    i32x8 z8 = {0, 0, 0, 0, 0, 0, 0, 0};
    __builtin_amdgcn_tensor_load_to_lds(g0, g1, z4, z4, z8, 0);
}

// ---------------------------------------------------------------------------
// Kernel 1: fused QKV projection.  y = x @ W^T  (M=B*T, K=EMB, N=HEAD)
// grid = M/16 blocks of 256 threads (8 waves).  One block computes the
// 16x128 output tiles of q, k AND v for its 16 rows (x staged once).
// ---------------------------------------------------------------------------
__global__ __launch_bounds__(256) void qkv_proj_kernel(
    const float* __restrict__ x,
    const float* __restrict__ Wq, const float* __restrict__ Wk,
    const float* __restrict__ Wv,
    _Float16* __restrict__ qb, _Float16* __restrict__ kb,
    _Float16* __restrict__ vb)
{
    __shared__ __align__(16) _Float16 xs[16 * 32];        // [row][k]
    __shared__ __align__(16) _Float16 wsT[3][32 * 128];   // [k][n] per matrix

    const int tid  = threadIdx.x;
    const int lane = tid & 31;
    const int wave = tid >> 5;               // N tile 0..7
    const int row0 = blockIdx.x * 16;
    const int rr   = lane & 15;
    const int hh   = lane >> 4;
    const float* Wm[3] = {Wq, Wk, Wv};

    v8f acc[3] = {{}, {}, {}};
    for (int kc = 0; kc < EMB; kc += 32) {
        // stage x chunk: 512 floats, 2/thread, vectorized read
        {
            int r = tid >> 4, k = (tid & 15) * 2;
            float2 x2 = *(const float2*)&x[(size_t)(row0 + r) * EMB + kc + k];
            xs[r * 32 + k]     = (_Float16)x2.x;
            xs[r * 32 + k + 1] = (_Float16)x2.y;
        }
        // stage W^T chunks: 4096 elems per matrix, 4x float4 per thread
        #pragma unroll
        for (int w = 0; w < 3; ++w) {
            const float* W = Wm[w];
            #pragma unroll
            for (int p = 0; p < 4; ++p) {
                int idx = p * 1024 + tid * 4;            // k fastest, k%4==0
                int n = idx >> 5, k = idx & 31;
                float4 w4 = *(const float4*)&W[(size_t)n * EMB + kc + k];
                wsT[w][(k + 0) * 128 + n] = (_Float16)w4.x;
                wsT[w][(k + 1) * 128 + n] = (_Float16)w4.y;
                wsT[w][(k + 2) * 128 + n] = (_Float16)w4.z;
                wsT[w][(k + 3) * 128 + n] = (_Float16)w4.w;
            }
        }
        __syncthreads();

        // A 16x32 f16: lane holds row rr; e=0..7 -> k=8*hh+e, e=8..15 -> k=16+8*hh+(e-8)
        v16h a;
        {
            const v8h lo = *(const v8h*)&xs[rr * 32 + 8 * hh];
            const v8h hi = *(const v8h*)&xs[rr * 32 + 16 + 8 * hh];
            #pragma unroll
            for (int e = 0; e < 8; ++e) { a[e] = lo[e]; a[8 + e] = hi[e]; }
        }
        #pragma unroll
        for (int w = 0; w < 3; ++w) {
            v16h b;                          // B 32x16: lane = n col, elem = k (table order)
            #pragma unroll
            for (int e = 0; e < 16; ++e)
                b[e] = wsT[w][((e & 8) * 2 + 8 * hh + (e & 7)) * 128 + wave * 16 + rr];
            acc[w] = __builtin_amdgcn_wmma_f32_16x16x32_f16(
                false, a, false, b, (short)0, acc[w], false, false);
        }
        __syncthreads();
    }

    #pragma unroll
    for (int r = 0; r < 8; ++r) {            // D layout: VGPR r -> M = r + 8*half
        size_t o = (size_t)(row0 + r + 8 * hh) * HEAD + wave * 16 + rr;
        qb[o] = (_Float16)(acc[0][r] * 0.08838834764831845f);  // 1/sqrt(128)
        kb[o] = (_Float16)acc[1][r];
        vb[o] = (_Float16)acc[2][r];
    }
}

// ---------------------------------------------------------------------------
// Kernel 2: causal flash attention, f16 q/k/v -> fp32 out.
// grid = (T/128, B), block = 256 (8 waves); each wave owns a 16-row Q tile.
// Q tile: TDM (tensor_load_to_lds, TENSORcnt).  V tiles: async b128 copies
// (ASYNCcnt).  K tiles: manual transposed staging.  K/V double-buffered.
// ---------------------------------------------------------------------------
__global__ __launch_bounds__(256) void flash_attn_kernel(
    const _Float16* __restrict__ qbuf, const _Float16* __restrict__ kbuf,
    const _Float16* __restrict__ vbuf, float* __restrict__ out)
{
    __shared__ __align__(16) _Float16 kT[2][128 * 32];       // [h][key]
    __shared__ __align__(16) _Float16 vs[2][32 * 128];       // [key][h]
    __shared__ __align__(16) _Float16 scratch[8][16 * 128];  // Q tile, then P

    const int tid  = threadIdx.x;
    const int lane = tid & 31;
    const int wave = tid >> 5;
    const int rr   = lane & 15;
    const int hh   = lane >> 4;

    // longest (largest qblock) workgroups first for causal load balance
    const int qblk  = (int)gridDim.x - 1 - (int)blockIdx.x;
    const int q0    = qblk * 128;
    const int qrow0 = q0 + wave * 16;
    const size_t base = (size_t)blockIdx.y * SEQ * HEAD;

    // --- Q tile via Tensor Data Mover: 16x128 f16 -> scratch[wave] ---
    tdm_load_tile_f16(lds_addr32(&scratch[wave][0]),
                      qbuf + base + (size_t)qrow0 * HEAD, 16u);

    auto stageK = [&](int kbase, int buf) {
        int key = tid >> 3;                  // 32 keys
        int h0  = (tid & 7) * 16;            // 8 h-groups of 16
        const _Float16* src = kbuf + base + (size_t)(kbase + key) * HEAD + h0;
        v8h k0 = *(const v8h*)src;
        v8h k1 = *(const v8h*)(src + 8);
        #pragma unroll
        for (int e = 0; e < 8; ++e) {
            kT[buf][(h0 + e) * 32 + key]     = k0[e];
            kT[buf][(h0 + 8 + e) * 32 + key] = k1[e];
        }
    };
    auto issueV = [&](int kbase, int buf) {  // 2x b128 async per thread
        const _Float16* src = vbuf + base + (size_t)kbase * HEAD + tid * 16;
        async_copy_b128(lds_addr32(&vs[buf][tid * 16]),     src);
        async_copy_b128(lds_addr32(&vs[buf][tid * 16 + 8]), src + 8);
    };

    stageK(0, 0);
    issueV(0, 0);

    // consume the TDM'd Q tile into A-layout registers
    __builtin_amdgcn_s_wait_tensorcnt((short)0);
    v16h qa[4];
    {
        const _Float16* qrow = &scratch[wave][rr * 128];
        #pragma unroll
        for (int c = 0; c < 4; ++c) {
            const v8h lo = *(const v8h*)(qrow + c * 32 + 8 * hh);
            const v8h hi = *(const v8h*)(qrow + c * 32 + 16 + 8 * hh);
            #pragma unroll
            for (int e = 0; e < 8; ++e) { qa[c][e] = lo[e]; qa[c][8 + e] = hi[e]; }
        }
    }
    _Float16* ps = &scratch[wave][0];        // reuse Q staging as P scratch

    v8f zero = {};
    v8f acc[8];
    #pragma unroll
    for (int t = 0; t < 8; ++t) acc[t] = zero;
    float m[8], lsum[8];
    #pragma unroll
    for (int r = 0; r < 8; ++r) { m[r] = -__builtin_inff(); lsum[r] = 0.0f; }

    const int kend = q0 + 128;               // causal bound (block-uniform)
    wait_async0();
    __syncthreads();                         // tile 0 resident

    for (int kbase = 0; kbase < kend; kbase += 32) {
        const int buf = (kbase >> 5) & 1;
        if (kbase + 32 < kend) {             // prefetch next tile into buf^1
            issueV(kbase + 32, buf ^ 1);
            stageK(kbase + 32, buf ^ 1);
        }

        if (kbase <= qrow0 + 15) {           // tile not fully masked (wave-uniform)
            // S = Q @ K^T : two 16x16 tiles
            v8f s[2]; s[0] = zero; s[1] = zero;
            #pragma unroll
            for (int c = 0; c < 4; ++c)
                #pragma unroll
                for (int n = 0; n < 2; ++n) {
                    v16h b;                  // B 32x16: lane = key col, elem = h (table order)
                    #pragma unroll
                    for (int e = 0; e < 16; ++e)
                        b[e] = kT[buf][(c * 32 + (e & 8) * 2 + 8 * hh + (e & 7)) * 32 + n * 16 + rr];
                    s[n] = __builtin_amdgcn_wmma_f32_16x16x32_f16(
                        false, qa[c], false, b, (short)0, s[n], false, false);
                }

            // causal mask + online softmax (row spans a 16-lane half)
            float alpha[8];
            #pragma unroll
            for (int r = 0; r < 8; ++r) {
                int qrow = qrow0 + r + 8 * hh;
                float s0 = s[0][r], s1 = s[1][r];
                if (kbase + rr      > qrow) s0 = -__builtin_inff();
                if (kbase + 16 + rr > qrow) s1 = -__builtin_inff();
                float rm = fmaxf(s0, s1);
                #pragma unroll
                for (int off = 1; off < 16; off <<= 1)
                    rm = fmaxf(rm, __shfl_xor(rm, off, 32));
                float mn = fmaxf(m[r], rm);
                alpha[r] = __expf(m[r] - mn);
                float p0 = __expf(s0 - mn);
                float p1 = __expf(s1 - mn);
                float rs = p0 + p1;
                #pragma unroll
                for (int off = 1; off < 16; off <<= 1)
                    rs += __shfl_xor(rs, off, 32);
                lsum[r] = lsum[r] * alpha[r] + rs;
                m[r] = mn;
                ps[(r + 8 * hh) * 32 + rr]      = (_Float16)p0;
                ps[(r + 8 * hh) * 32 + 16 + rr] = (_Float16)p1;
            }
            asm volatile("s_wait_dscnt 0x0" ::: "memory");  // wave-local LDS RAW

            // reload P in A layout (vectorized: contiguous 16B per half)
            v16h pa;
            {
                const v8h lo = *(const v8h*)&ps[rr * 32 + 8 * hh];
                const v8h hi = *(const v8h*)&ps[rr * 32 + 16 + 8 * hh];
                #pragma unroll
                for (int e = 0; e < 8; ++e) { pa[e] = lo[e]; pa[8 + e] = hi[e]; }
            }

            // O = O*alpha + P @ V
            #pragma unroll
            for (int t = 0; t < 8; ++t) {
                v8f c2;
                #pragma unroll
                for (int r = 0; r < 8; ++r) c2[r] = acc[t][r] * alpha[r];
                v16h b;                      // B 32x16: lane = h col, elem = key (table order)
                #pragma unroll
                for (int e = 0; e < 16; ++e)
                    b[e] = vs[buf][((e & 8) * 2 + 8 * hh + (e & 7)) * 128 + t * 16 + rr];
                acc[t] = __builtin_amdgcn_wmma_f32_16x16x32_f16(
                    false, pa, false, b, (short)0, c2, false, false);
            }
        }

        wait_async0();                       // next V tile landed
        __syncthreads();                     // flush K stores, join waves
    }

    float inv[8];
    #pragma unroll
    for (int r = 0; r < 8; ++r) inv[r] = 1.0f / lsum[r];
    #pragma unroll
    for (int t = 0; t < 8; ++t)
        #pragma unroll
        for (int r = 0; r < 8; ++r)
            out[base + (size_t)(qrow0 + r + 8 * hh) * HEAD + t * 16 + rr] =
                acc[t][r] * inv[r];
}

// ---------------------------------------------------------------------------
extern "C" void kernel_launch(void* const* d_in, const int* in_sizes, int n_in,
                              void* d_out, int out_size, void* d_ws, size_t ws_size,
                              hipStream_t stream) {
    const float* x  = (const float*)d_in[0];
    const float* Wq = (const float*)d_in[1];
    const float* Wk = (const float*)d_in[2];
    const float* Wv = (const float*)d_in[3];

    const size_t n = (size_t)BATCH * SEQ * HEAD;   // ws: 24 MB of f16 q,k,v
    _Float16* qb = (_Float16*)d_ws;
    _Float16* kb = qb + n;
    _Float16* vb = kb + n;

    qkv_proj_kernel<<<(BATCH * SEQ) / 16, 256, 0, stream>>>(x, Wq, Wk, Wv, qb, kb, vb);

    dim3 g2(SEQ / 128, BATCH);
    flash_attn_kernel<<<g2, 256, 0, stream>>>(qb, kb, vb, (float*)d_out);
}
